// MultiHeadAttention_57741540327756
// MI455X (gfx1250) — matrix operand order repaired and checked by reading, not compiled, not run on hardware
//
#include <hip/hip_runtime.h>
#include <math.h>

// ---------------------------------------------------------------------------
// Types for CDNA5 WMMA (wave32, 16x16x32 bf16 -> f32)
// ---------------------------------------------------------------------------
typedef __bf16 bf16_t;
typedef __attribute__((ext_vector_type(16))) __bf16 v16bf;
typedef __attribute__((ext_vector_type(8)))  __bf16 v8bf;
typedef __attribute__((ext_vector_type(8)))  float  v8f;

struct bfpair { v8bf lo, hi; };

static __device__ __forceinline__ v16bf make_frag(v8bf lo, v8bf hi) {
  bfpair c{lo, hi};
  return __builtin_bit_cast(v16bf, c);
}

// A fragment: 16x32 bf16 tile (MxK), row-major memory, leading dim lda.
// ISA layout: lane m=l%16; elements 0..7 hold K = 8*(l/16)+0..7,
// elements 8..15 hold K = 16+8*(l/16)+0..7  -> two contiguous 16B loads.
static __device__ __forceinline__ v16bf load_a_frag(const bf16_t* A, int lda) {
  int lane = threadIdx.x & 31;
  int m = lane & 15, hf = lane >> 4;
  const bf16_t* p = A + (size_t)m * lda + 8 * hf;
  v8bf lo = *(const v8bf*)p;
  v8bf hi = *(const v8bf*)(p + 16);
  return make_frag(lo, hi);
}

// B fragment: 32x16 (KxN) where memory is N-major: element(k,n) at W[n*ldw+k].
// ISA layout: lane n=l%16; elements 0..7 hold K = 8*(l/16)+0..7,
// elements 8..15 hold K = 16+8*(l/16)+0..7  -> two contiguous 16B loads.
static __device__ __forceinline__ v16bf load_b_frag(const bf16_t* W, int ldw) {
  int lane = threadIdx.x & 31;
  int n = lane & 15, hf = lane >> 4;
  const bf16_t* p = W + (size_t)n * ldw + 8 * hf;
  v8bf lo = *(const v8bf*)p;
  v8bf hi = *(const v8bf*)(p + 16);
  return make_frag(lo, hi);
}

#define WMMA_BF16(a, b, c) \
  __builtin_amdgcn_wmma_f32_16x16x32_bf16(false, (a), false, (b), (short)0, (c), false, false)

// ---------------------------------------------------------------------------
// fp32 -> bf16 bulk convert
// ---------------------------------------------------------------------------
__global__ void k_f32_to_bf16(const float* __restrict__ in, bf16_t* __restrict__ out, size_t n) {
  size_t i = (size_t)blockIdx.x * blockDim.x + threadIdx.x;
  if (i < n) out[i] = (bf16_t)in[i];
}

// ---------------------------------------------------------------------------
// pos = cumsum(mask) - 1
// ---------------------------------------------------------------------------
__global__ void k_pos(const int* __restrict__ mask, int* __restrict__ pos, int B, int T) {
  int b = blockIdx.x * blockDim.x + threadIdx.x;
  if (b >= B) return;
  int c = 0;
  for (int t = 0; t < T; ++t) { c += mask[b * T + t]; pos[b * T + t] = c - 1; }
}

// ---------------------------------------------------------------------------
// C = A(MxK) @ W(NxK)^T  [+ bias], bf16 in, f32 acc. Software-pipelined.
// Wave computes 32x64 of C (2 A-frags x 4 B-frags = 8 WMMA / 32-k step).
// Block = 4 waves (128x64 of C); grid = (N/64, M/128).
// __launch_bounds__(128,1): let the allocator keep all 8 acc frags +
// double-buffered operand frags resident (~192 VGPRs) with zero spill.
// ---------------------------------------------------------------------------
__global__ void __launch_bounds__(128, 1)
k_gemm_nt_bf16(const bf16_t* __restrict__ A,
               const bf16_t* __restrict__ W,
               const float*  __restrict__ bias,   // nullable
               float*        __restrict__ Cf,     // nullable
               bf16_t*       __restrict__ Cb,     // nullable
               int M, int N, int K) {
  int wave = threadIdx.x >> 5;
  int lane = threadIdx.x & 31;
  int m0 = blockIdx.y * 128 + wave * 32;
  int n0 = blockIdx.x * 64;

  const bf16_t* Ar0 = A + (size_t)m0 * K;
  const bf16_t* Ar1 = A + (size_t)(m0 + 16) * K;
  const bf16_t* W0 = W + (size_t)(n0)      * K;
  const bf16_t* W1 = W + (size_t)(n0 + 16) * K;
  const bf16_t* W2 = W + (size_t)(n0 + 32) * K;
  const bf16_t* W3 = W + (size_t)(n0 + 48) * K;

  v8f acc0[4] = {}, acc1[4] = {};

  // prologue: load k-step 0
  v16bf a0 = load_a_frag(Ar0, K);
  v16bf a1 = load_a_frag(Ar1, K);
  v16bf b0 = load_b_frag(W0, K);
  v16bf b1 = load_b_frag(W1, K);
  v16bf b2 = load_b_frag(W2, K);
  v16bf b3 = load_b_frag(W3, K);

  for (int k0 = 32; k0 < K; k0 += 32) {
    // issue next k-step loads (overlap with 8 WMMAs below)
    v16bf na0 = load_a_frag(Ar0 + k0, K);
    v16bf na1 = load_a_frag(Ar1 + k0, K);
    v16bf nb0 = load_b_frag(W0 + k0, K);
    v16bf nb1 = load_b_frag(W1 + k0, K);
    v16bf nb2 = load_b_frag(W2 + k0, K);
    v16bf nb3 = load_b_frag(W3 + k0, K);

    acc0[0] = WMMA_BF16(a0, b0, acc0[0]);
    acc1[0] = WMMA_BF16(a1, b0, acc1[0]);
    acc0[1] = WMMA_BF16(a0, b1, acc0[1]);
    acc1[1] = WMMA_BF16(a1, b1, acc1[1]);
    acc0[2] = WMMA_BF16(a0, b2, acc0[2]);
    acc1[2] = WMMA_BF16(a1, b2, acc1[2]);
    acc0[3] = WMMA_BF16(a0, b3, acc0[3]);
    acc1[3] = WMMA_BF16(a1, b3, acc1[3]);

    a0 = na0; a1 = na1; b0 = nb0; b1 = nb1; b2 = nb2; b3 = nb3;
  }
  // epilogue k-step
  acc0[0] = WMMA_BF16(a0, b0, acc0[0]);
  acc1[0] = WMMA_BF16(a1, b0, acc1[0]);
  acc0[1] = WMMA_BF16(a0, b1, acc0[1]);
  acc1[1] = WMMA_BF16(a1, b1, acc1[1]);
  acc0[2] = WMMA_BF16(a0, b2, acc0[2]);
  acc1[2] = WMMA_BF16(a1, b2, acc1[2]);
  acc0[3] = WMMA_BF16(a0, b3, acc0[3]);
  acc1[3] = WMMA_BF16(a1, b3, acc1[3]);

  int nl = lane & 15, hf = lane >> 4;
#pragma unroll
  for (int t = 0; t < 4; ++t) {
#pragma unroll
    for (int v = 0; v < 8; ++v) {
      int n = n0 + 16 * t + nl;
      float bv = bias ? bias[n] : 0.0f;
      int ma = m0 + v + 8 * hf;
      int mb = ma + 16;
      float va = acc0[t][v] + bv;
      float vb = acc1[t][v] + bv;
      if (Cf) { Cf[(size_t)ma * N + n] = va; Cf[(size_t)mb * N + n] = vb; }
      if (Cb) { Cb[(size_t)ma * N + n] = (bf16_t)va; Cb[(size_t)mb * N + n] = (bf16_t)vb; }
    }
  }
}

// ---------------------------------------------------------------------------
// gate = sigmoid(x @ Wgate^T)   (B*T, h) — tiny GEMV
// ---------------------------------------------------------------------------
__global__ void k_gate(const float* __restrict__ x, const float* __restrict__ Wg,
                       float* __restrict__ gate, int BT, int D, int h) {
  int idx = blockIdx.x * blockDim.x + threadIdx.x;
  if (idx >= BT * h) return;
  int head = idx % h, bt = idx / h;
  const float4* xr = (const float4*)(x + (size_t)bt * D);
  const float4* wr = (const float4*)(Wg + (size_t)head * D);
  float s = 0.f;
  for (int k = 0; k < D / 4; ++k) {
    float4 a = xr[k], b = wr[k];
    s += a.x * b.x + a.y * b.y + a.z * b.z + a.w * b.w;
  }
  gate[idx] = 1.f / (1.f + __expf(-s));
}

// ---------------------------------------------------------------------------
// RoPE on Q,K (bf16 in, (B,T,D)) fused with relayout -> (B,h,T,H) bf16.
// ---------------------------------------------------------------------------
__global__ void k_rope(const bf16_t* __restrict__ q, const bf16_t* __restrict__ k,
                       const int* __restrict__ pos,
                       bf16_t* __restrict__ qo, bf16_t* __restrict__ ko,
                       int B, int T, int h, int H) {
  int half = H / 2;
  int idx = blockIdx.x * blockDim.x + threadIdx.x;
  int total = B * T * h * half;
  if (idx >= total) return;
  int i = idx % half; int tmp = idx / half;
  int head = tmp % h; tmp /= h;
  int t = tmp % T; int b = tmp / T;

  float p = (float)pos[b * T + t];
  float inv_freq = __expf(-((float)(2 * i) / (float)H) * logf(10000.0f));
  float ang = p * inv_freq;
  float c = cosf(ang), s = sinf(ang);

  size_t src = ((size_t)(b * T + t) * h + head) * H;
  float q0 = (float)q[src + i], q1 = (float)q[src + i + half];
  float k0 = (float)k[src + i], k1 = (float)k[src + i + half];

  size_t dst = ((size_t)(b * h + head) * T + t) * (size_t)H;
  qo[dst + i]        = (bf16_t)(q0 * c - q1 * s);
  qo[dst + i + half] = (bf16_t)(q1 * c + q0 * s);
  ko[dst + i]        = (bf16_t)(k0 * c - k1 * s);
  ko[dst + i + half] = (bf16_t)(k1 * c + k0 * s);
}

// ---------------------------------------------------------------------------
// V relayout (B,T,D) bf16 -> Vt (B,h,H,T) bf16  (transposed for P@V B-frags)
// ---------------------------------------------------------------------------
__global__ void k_vtrans(const bf16_t* __restrict__ v, bf16_t* __restrict__ vt,
                         int B, int T, int h, int H) {
  size_t idx = (size_t)blockIdx.x * blockDim.x + threadIdx.x;
  size_t total = (size_t)B * h * H * T;
  if (idx >= total) return;
  int t = idx % T; size_t tmp = idx / T;
  int hd = tmp % H; tmp /= H;
  int head = tmp % h; int b = tmp / h;
  vt[idx] = v[((size_t)(b * T + t) * h + head) * H + hd];
}

// ---------------------------------------------------------------------------
// Flash attention: one wave per 16 query rows, online softmax, 32 keys/iter.
// Q,K: (B,h,T,H) bf16; Vt: (B,h,H,T) bf16. Output gated bf16 (B,T,D).
// ---------------------------------------------------------------------------
#define ATTN_WAVES 8
__global__ void __launch_bounds__(32 * ATTN_WAVES, 1)
k_attn(const bf16_t* __restrict__ Q, const bf16_t* __restrict__ Km,
       const bf16_t* __restrict__ Vt, const int* __restrict__ mask,
       const float* __restrict__ gate, bf16_t* __restrict__ Ob,
       int B, int T, int h, int H) {
  __shared__ float lds[ATTN_WAVES][16][33];   // per-wave P bounce (C-layout -> A-layout)

  int wave = threadIdx.x >> 5, lane = threadIdx.x & 31;
  int nl = lane & 15, hf = lane >> 4;
  int head = blockIdx.y, b = blockIdx.z;
  int q0 = (blockIdx.x * ATTN_WAVES + wave) * 16;

  const bf16_t* Qh = Q  + ((size_t)(b * h + head) * T) * H;
  const bf16_t* Kh = Km + ((size_t)(b * h + head) * T) * H;
  const bf16_t* Vh = Vt + ((size_t)(b * h + head) * H) * T;

  v16bf qa0 = load_a_frag(Qh + (size_t)q0 * H, H);        // feats 0..31
  v16bf qa1 = load_a_frag(Qh + (size_t)q0 * H + 32, H);   // feats 32..63

  v8f acc[4] = {};
  float rowm[8], rowl[8];
#pragma unroll
  for (int v = 0; v < 8; ++v) { rowm[v] = -3.402823466e38f; rowl[v] = 0.f; }

  const float scale = rsqrtf((float)H);
  const float NEG = -3.402823466e38f;
  int kend = q0 + 16;                                     // causal: keys <= q0+15

  for (int kt = 0; kt < kend; kt += 32) {
    // ---- issue all loads for this step up front (K frags + V frags) ----
    const bf16_t* Kb0 = Kh + (size_t)kt * H;
    const bf16_t* Kb1 = Kh + (size_t)(kt + 16) * H;
    v16bf kb00 = load_b_frag(Kb0, H);        // keys kt..+15, feat 0..31
    v16bf kb01 = load_b_frag(Kb0 + 32, H);   // feat 32..63
    v16bf kb10 = load_b_frag(Kb1, H);
    v16bf kb11 = load_b_frag(Kb1 + 32, H);
    v16bf vb[4];
#pragma unroll
    for (int t = 0; t < 4; ++t)
      vb[t] = load_b_frag(Vh + (size_t)(16 * t) * T + kt, T);   // V frags overlap softmax

    // ---- S = Q @ K^T ----
    v8f s[2] = {};
    s[0] = WMMA_BF16(qa0, kb00, s[0]);
    s[0] = WMMA_BF16(qa1, kb01, s[0]);
    s[1] = WMMA_BF16(qa0, kb10, s[1]);
    s[1] = WMMA_BF16(qa1, kb11, s[1]);

    // ---- scale + causal/padding mask ----
#pragma unroll
    for (int st = 0; st < 2; ++st) {
      int kcol = kt + 16 * st + nl;
      int mk = (kcol < T) ? mask[b * T + kcol] : 0;
#pragma unroll
      for (int v = 0; v < 8; ++v) {
        int qrow = q0 + v + 8 * hf;
        s[st][v] = (mk != 0 && kcol <= qrow) ? s[st][v] * scale : NEG;
      }
    }
    // ---- online softmax ----
    float fact[8];
#pragma unroll
    for (int v = 0; v < 8; ++v) {
      float mx = fmaxf(s[0][v], s[1][v]);
#pragma unroll
      for (int off = 1; off < 16; off <<= 1) mx = fmaxf(mx, __shfl_xor(mx, off, 32));
      float nm = fmaxf(rowm[v], mx);
      fact[v] = __expf(rowm[v] - nm);
      rowm[v] = nm;
      float p0 = __expf(s[0][v] - nm);
      float p1 = __expf(s[1][v] - nm);
      s[0][v] = p0; s[1][v] = p1;
      float rs = p0 + p1;
#pragma unroll
      for (int off = 1; off < 16; off <<= 1) rs += __shfl_xor(rs, off, 32);
      rowl[v] = rowl[v] * fact[v] + rs;
    }
#pragma unroll
    for (int t = 0; t < 4; ++t)
#pragma unroll
      for (int v = 0; v < 8; ++v) acc[t][v] = acc[t][v] * fact[v];

    // ---- P: C-layout -> A-layout via per-wave LDS (wave-synchronous) ----
#pragma unroll
    for (int st = 0; st < 2; ++st)
#pragma unroll
      for (int v = 0; v < 8; ++v)
        lds[wave][v + 8 * hf][16 * st + nl] = s[st][v];
    __builtin_amdgcn_wave_barrier();
    v8bf plo, phi;
#pragma unroll
    for (int e = 0; e < 8; ++e) {
      plo[e] = (bf16_t)lds[wave][nl][8 * hf + e];
      phi[e] = (bf16_t)lds[wave][nl][16 + 8 * hf + e];
    }
    __builtin_amdgcn_wave_barrier();
    v16bf pa = make_frag(plo, phi);

    // ---- O += P @ V ----
#pragma unroll
    for (int t = 0; t < 4; ++t) acc[t] = WMMA_BF16(pa, vb[t], acc[t]);
  }

  // ---- epilogue: 1/rowsum, per-head gate, store bf16 (B,T,D) ----
#pragma unroll
  for (int v = 0; v < 8; ++v) {
    int qrow = q0 + v + 8 * hf;
    float g = gate[((size_t)b * T + qrow) * h + head];
    float inv = (rowl[v] > 0.f) ? (g / rowl[v]) : 0.f;
#pragma unroll
    for (int t = 0; t < 4; ++t) {
      int feat = 16 * t + nl;
      Ob[((size_t)(b * T + qrow) * h + head) * H + feat] = (bf16_t)(acc[t][v] * inv);
    }
  }
}

// ---------------------------------------------------------------------------
// Host-side launcher
// ---------------------------------------------------------------------------
static inline size_t align_up(size_t x, size_t a) { return (x + a - 1) & ~(a - 1); }

extern "C" void kernel_launch(void* const* d_in, const int* in_sizes, int n_in,
                              void* d_out, int out_size, void* d_ws, size_t ws_size,
                              hipStream_t stream) {
  const float* x     = (const float*)d_in[0];
  const int*   mask  = (const int*)  d_in[1];
  const float* Wq    = (const float*)d_in[2];
  const float* Wk    = (const float*)d_in[3];
  const float* Wv    = (const float*)d_in[4];
  const float* Wgate = (const float*)d_in[5];
  const float* Wproj = (const float*)d_in[6];
  const float* bproj = (const float*)d_in[7];
  float* out = (float*)d_out;

  // Shapes (reference: B=2, T=2048, D=1024, h=16)
  const int D  = in_sizes[7];            // bproj is (D,)
  const int h  = in_sizes[5] / D;        // Wgate is (h, D)
  const int BT = in_sizes[0] / D;        // x is (B*T, D)
  const int T  = 2048;
  const int B  = BT / T;
  const int H  = D / h;

  // Workspace carve-up
  char* ws = (char*)d_ws;
  size_t off = 0;
  auto alloc = [&](size_t bytes) { char* p = ws + off; off = align_up(off + bytes, 256); return p; };
  bf16_t* xb   = (bf16_t*)alloc((size_t)BT * D * 2);
  bf16_t* Wqb  = (bf16_t*)alloc((size_t)D * D * 2);
  bf16_t* Wkb  = (bf16_t*)alloc((size_t)D * D * 2);
  bf16_t* Wvb  = (bf16_t*)alloc((size_t)D * D * 2);
  bf16_t* Wpb  = (bf16_t*)alloc((size_t)D * D * 2);
  bf16_t* qp   = (bf16_t*)alloc((size_t)BT * D * 2);   // (B,T,D) pre-RoPE
  bf16_t* kp   = (bf16_t*)alloc((size_t)BT * D * 2);
  bf16_t* vp   = (bf16_t*)alloc((size_t)BT * D * 2);
  bf16_t* qb   = (bf16_t*)alloc((size_t)BT * D * 2);   // (B,h,T,H) post-RoPE
  bf16_t* kb   = (bf16_t*)alloc((size_t)BT * D * 2);
  bf16_t* vtb  = (bf16_t*)alloc((size_t)BT * D * 2);   // (B,h,H,T)
  bf16_t* ob   = (bf16_t*)alloc((size_t)BT * D * 2);   // gated attn out (B,T,D)
  float*  gate = (float*) alloc((size_t)BT * h * 4);
  int*    pos  = (int*)   alloc((size_t)BT * 4);

  const int CT = 256;
  // 1) fp32 -> bf16
  size_t nx = (size_t)BT * D, nw = (size_t)D * D;
  k_f32_to_bf16<<<dim3((nx + CT - 1) / CT), CT, 0, stream>>>(x, xb, nx);
  k_f32_to_bf16<<<dim3((nw + CT - 1) / CT), CT, 0, stream>>>(Wq, Wqb, nw);
  k_f32_to_bf16<<<dim3((nw + CT - 1) / CT), CT, 0, stream>>>(Wk, Wkb, nw);
  k_f32_to_bf16<<<dim3((nw + CT - 1) / CT), CT, 0, stream>>>(Wv, Wvb, nw);
  k_f32_to_bf16<<<dim3((nw + CT - 1) / CT), CT, 0, stream>>>(Wproj, Wpb, nw);

  // 2) positions from padding mask
  k_pos<<<1, 32, 0, stream>>>(mask, pos, B, T);

  // 3) Q/K/V projections (WMMA bf16, pipelined) -> bf16 (B,T,D)
  dim3 ggrid(D / 64, BT / 128);
  k_gemm_nt_bf16<<<ggrid, 128, 0, stream>>>(xb, Wqb, nullptr, nullptr, qp, BT, D, D);
  k_gemm_nt_bf16<<<ggrid, 128, 0, stream>>>(xb, Wkb, nullptr, nullptr, kp, BT, D, D);
  k_gemm_nt_bf16<<<ggrid, 128, 0, stream>>>(xb, Wvb, nullptr, nullptr, vp, BT, D, D);

  // 4) per-head sigmoid gate
  k_gate<<<dim3((BT * h + CT - 1) / CT), CT, 0, stream>>>(x, Wgate, gate, BT, D, h);

  // 5) RoPE + relayout Q,K -> (B,h,T,H)
  int nrope = B * T * h * (H / 2);
  k_rope<<<dim3((nrope + CT - 1) / CT), CT, 0, stream>>>(qp, kp, pos, qb, kb, B, T, h, H);

  // 6) V -> (B,h,H,T)
  size_t nvt = (size_t)B * h * H * T;
  k_vtrans<<<dim3((nvt + CT - 1) / CT), CT, 0, stream>>>(vp, vtb, B, T, h, H);

  // 7) flash attention with causal+padding mask, gated epilogue -> bf16 (B,T,D)
  dim3 agrid(T / (16 * ATTN_WAVES), h, B);
  k_attn<<<agrid, 32 * ATTN_WAVES, 0, stream>>>(qb, kb, vtb, mask, gate, ob, B, T, h, H);

  // 8) output projection + bias -> f32 d_out
  k_gemm_nt_bf16<<<ggrid, 128, 0, stream>>>(ob, Wpb, bproj, out, nullptr, BT, D, D);
}
